// MultiHeadAttention_9448928051627
// MI455X (gfx1250) — matrix operand order repaired and checked by reading, not compiled, not run on hardware
//
#include <hip/hip_runtime.h>
#include <hip/hip_bf16.h>

// MHA forward for B=4, S=2048, D=1024, H=16, HD=64 on gfx1250 (MI455X).
// All matmuls on V_WMMA_F32_16X16X32_BF16 (f32 accum), wave32.
// TDM (tensor_load_to_lds) stages pre-converted bf16 tiles where possible.

#define BB 4
#define SS 2048
#define DD 1024
#define HH 16
#define HDIM 64

typedef __attribute__((ext_vector_type(16))) __bf16 v16bf;
typedef __attribute__((ext_vector_type(8)))  float  v8f;
typedef unsigned int u32x4 __attribute__((ext_vector_type(4)));
typedef int          i32x4 __attribute__((ext_vector_type(4)));
typedef int          i32x8 __attribute__((ext_vector_type(8)));

#if __has_builtin(__builtin_amdgcn_tensor_load_to_lds) && \
    __has_builtin(__builtin_amdgcn_s_wait_tensorcnt)
#define USE_TDM 1
#else
#define USE_TDM 0
#endif

union BFVec { v16bf v; unsigned short u[16]; };

__device__ __forceinline__ unsigned short f2bf(float f) {
  unsigned int x = __float_as_uint(f);
  x += 0x7FFFu + ((x >> 16) & 1u);          // round-to-nearest-even
  return (unsigned short)(x >> 16);
}

// Packed fp32x2 -> bf16x2 (low = lo, high = hi) as one u32.
#if __has_builtin(__builtin_amdgcn_cvt_pk_bf16_f32)
__device__ __forceinline__ unsigned f2bf_pk(float lo, float hi) {
  typedef __bf16 v2bf __attribute__((ext_vector_type(2)));
  union { v2bf v; unsigned u; } c;
  c.v = __builtin_amdgcn_cvt_pk_bf16_f32(lo, hi);   // v_cvt_pk_bf16_f32
  return c.u;
}
#else
__device__ __forceinline__ unsigned f2bf_pk(float lo, float hi) {
  return (unsigned)f2bf(lo) | ((unsigned)f2bf(hi) << 16);
}
#endif

// A-matrix fragment (16x32 bf16): lane holds row (lane&15); half = lane>>4.
// elements 0..7  <-> k = half*8 + e ; elements 8..15 <-> k = 16 + half*8 + e
__device__ __forceinline__ v16bf load_fragA(const unsigned short* rowp, int half) {
  BFVec f;
  const unsigned short* p0 = rowp + half * 8;
  const unsigned short* p1 = rowp + 16 + half * 8;
#pragma unroll
  for (int e = 0; e < 8; ++e) { f.u[e] = p0[e]; f.u[8 + e] = p1[e]; }
  return f.v;
}

// B-matrix fragment (32x16 bf16): lane holds column (lane&15);
// elements 0..7  <-> k = half*8 + e ; elements 8..15 <-> k = 16 + half*8 + e (colp = &lds[n][0])
__device__ __forceinline__ v16bf load_fragB(const unsigned short* colp, int half) {
  BFVec f;
  const unsigned short* p0 = colp + half * 8;
  const unsigned short* p1 = colp + 16 + half * 8;
#pragma unroll
  for (int e = 0; e < 8; ++e) { f.u[e] = p0[e]; f.u[8 + e] = p1[e]; }
  return f.v;
}

__device__ __forceinline__ v8f wmma_bf16(v16bf a, v16bf b, v8f c) {
  return __builtin_amdgcn_wmma_f32_16x16x32_bf16(false, a, false, b, (short)0, c,
                                                 false, false);
}

#if USE_TDM
// 2D TDM load of bf16 tile (data_size=2B) with LDS row padding.
// D# bit layout per CDNA5 ISA ch.8: group0 = {count/type/addr}, group1 = dims.
// pad_interval_code: pad after 2^(c+1) dwords; pad_amount_code: pad (c+1) dwords.
// This toolchain (clang-23 lane) declares the 6-arg builtin:
//   (u32x4 g0, i32x8 g1, i32x4 g2, i32x4 g3, i32x8 extra, i32 cpol)
__device__ __forceinline__ void tdm_load_2d_bf16(
    const unsigned short* lds_dst, const unsigned short* gsrc,
    unsigned tile_w, unsigned tile_h, unsigned row_stride_elems,
    unsigned pad_interval_code, unsigned pad_amount_code)
{
  const unsigned long long ga = (unsigned long long)(size_t)gsrc;
  u32x4 g0;
  g0[0] = 1u;                                          // count=1, user mode
  g0[1] = (unsigned)(size_t)lds_dst;                   // lds_addr (bytes)
  g0[2] = (unsigned)ga;                                // global_addr[31:0]
  g0[3] = ((unsigned)(ga >> 32) & 0x01FFFFFFu)         // global_addr[56:32]
          | 0x80000000u;                               // type = 2 ("image")
  i32x8 g1;
  g1[0] = (int)((1u << 16)                             // data_size = 2B
              | (1u << 20)                             // pad_enable
              | (pad_interval_code << 22)
              | (pad_amount_code << 25));              // workgroup_mask = 0
  g1[1] = (int)(tile_w << 16);                         // tensor_dim0 lo16
  g1[2] = (int)((tile_w >> 16) | (tile_h << 16));      // tensor_dim0 hi / dim1 lo
  g1[3] = (int)((tile_h >> 16) | (tile_w << 16));      // dim1 hi / tile_dim0
  g1[4] = (int)(tile_h & 0xFFFFu);                     // tile_dim1 (tile_dim2=0)
  g1[5] = (int)row_stride_elems;                       // tensor_dim0_stride lo32
  g1[6] = 0;                                           // stride hi / dim1_stride
  g1[7] = 0;
  const i32x4 z4 = {0, 0, 0, 0};                       // groups 2/3 unused (2D)
  const i32x8 z8 = {0, 0, 0, 0, 0, 0, 0, 0};
  __builtin_amdgcn_tensor_load_to_lds(g0, g1, z4, z4, z8, 0);
}
#endif

// ---------------------------------------------------------------------------
// GEMM: workgroup tile 128x128, 8 waves of 32x64 (2x4 wmma frags), K-step 32.
// ---------------------------------------------------------------------------
#define P_TM   128
#define P_TN   128
#define P_TK   32
#define P_ASTR 40   // bf16 per LDS A row (32 + 8 pad; 80B rows, 16B aligned)
#define P_BSTR 40   // bf16 per LDS B row ([n][k], padded)

// X[8192,1024] fp32 @ W[1024,1024] fp32 + b, scaled -> bf16 head-major [B,H,S,HD]
__global__ void __launch_bounds__(256) qkv_proj_kernel(
    const float* __restrict__ X, const float* __restrict__ W,
    const float* __restrict__ bias, unsigned short* __restrict__ dst, float scale)
{
  __shared__ unsigned short lsA[P_TM * P_ASTR];
  __shared__ unsigned short lsB[P_TN * P_BSTR];

  const int t    = threadIdx.x;
  const int mb   = blockIdx.x * P_TM;
  const int nb   = blockIdx.y * P_TN;
  const int wave = t >> 5, lane = t & 31;
  const int wm   = wave & 3, wn = wave >> 2;          // 4 x 2 wave grid
  const int half = lane >> 4, r = lane & 15;

  v8f acc[2][4] = {};

  for (int kk = 0; kk < DD; kk += P_TK) {
    __syncthreads();
    // Stage A tile: 128x32 fp32 -> bf16 via v_cvt_pk_bf16_f32, b64 LDS stores
#pragma unroll
    for (int i = 0; i < 4; ++i) {
      const int id  = t + i * 256;                    // 1024 float4 chunks
      const int row = id >> 3;
      const int cg  = id & 7;
      const float4 f = *(const float4*)(X + (size_t)(mb + row) * DD + kk + cg * 4);
      const unsigned long long pk =
          (unsigned long long)f2bf_pk(f.x, f.y) |
          ((unsigned long long)f2bf_pk(f.z, f.w) << 32);
      *(unsigned long long*)&lsA[row * P_ASTR + cg * 4] = pk;
    }
    // Stage B tile transposed: W rows kk..kk+31, cols nb..nb+127 -> lsB[n][k]
#pragma unroll
    for (int i = 0; i < 4; ++i) {
      const int id   = t + i * 256;                   // 1024 float4 chunks
      const int krow = id >> 5;
      const int ncg  = id & 31;
      const float4 f = *(const float4*)(W + (size_t)(kk + krow) * DD + nb + ncg * 4);
      const unsigned p01 = f2bf_pk(f.x, f.y);
      const unsigned p23 = f2bf_pk(f.z, f.w);
      lsB[(ncg * 4 + 0) * P_BSTR + krow] = (unsigned short)p01;
      lsB[(ncg * 4 + 1) * P_BSTR + krow] = (unsigned short)(p01 >> 16);
      lsB[(ncg * 4 + 2) * P_BSTR + krow] = (unsigned short)p23;
      lsB[(ncg * 4 + 3) * P_BSTR + krow] = (unsigned short)(p23 >> 16);
    }
    __syncthreads();

    v16bf af[2], bfr[4];
#pragma unroll
    for (int fm = 0; fm < 2; ++fm)
      af[fm] = load_fragA(&lsA[(wm * 32 + fm * 16 + r) * P_ASTR], half);
#pragma unroll
    for (int fn = 0; fn < 4; ++fn)
      bfr[fn] = load_fragB(&lsB[(wn * 64 + fn * 16 + r) * P_BSTR], half);
#pragma unroll
    for (int fm = 0; fm < 2; ++fm)
#pragma unroll
      for (int fn = 0; fn < 4; ++fn)
        acc[fm][fn] = wmma_bf16(af[fm], bfr[fn], acc[fm][fn]);
  }

  // Write back: bias + scale, bf16, head-major [B,H,S,HD].
#pragma unroll
  for (int fm = 0; fm < 2; ++fm) {
#pragma unroll
    for (int fn = 0; fn < 4; ++fn) {
      const int col = nb + wn * 64 + fn * 16 + r;
      const int h   = col >> 6, hd = col & 63;
      const float bc = bias[col];
#pragma unroll
      for (int j = 0; j < 8; ++j) {
        const int row = mb + wm * 32 + fm * 16 + half * 8 + j;
        const int b_  = row >> 11, s_ = row & 2047;
        dst[((size_t)(b_ * HH + h) * SS + s_) * HDIM + hd] =
            f2bf((acc[fm][fn][j] + bc) * scale);
      }
    }
  }
}

// O head-major bf16 @ Wo fp32 + bo -> fp32 [B,S,D]
__global__ void __launch_bounds__(256) out_proj_kernel(
    const unsigned short* __restrict__ Aws, const float* __restrict__ W,
    const float* __restrict__ bias, float* __restrict__ out)
{
  __shared__ unsigned short lsA[P_TM * P_ASTR];
  __shared__ unsigned short lsB[P_TN * P_BSTR];

  const int t    = threadIdx.x;
  const int mb   = blockIdx.x * P_TM;
  const int nb   = blockIdx.y * P_TN;
  const int wave = t >> 5, lane = t & 31;
  const int wm   = wave & 3, wn = wave >> 2;
  const int half = lane >> 4, r = lane & 15;
  const int b0   = mb >> 11, s0 = mb & 2047;          // 128-row block: one batch

  v8f acc[2][4] = {};

  for (int kk = 0; kk < DD; kk += P_TK) {
    __syncthreads();
    const int h  = kk >> 6;                           // K-step stays in one head
    const int hk = kk & 63;
#if USE_TDM
    // A tile is already bf16 & contiguous rows of 32 -> one TDM DMA,
    // hardware pads 8 bf16 per 32-element row (stride 40 = P_ASTR).
    if (wave == 0)
      tdm_load_2d_bf16(lsA,
                       Aws + ((size_t)(b0 * HH + h) * SS + s0) * HDIM + hk,
                       /*tile_w=*/32, /*tile_h=*/128, /*row_stride=*/HDIM,
                       /*pad_interval(16dw)=*/3, /*pad_amount(4dw)=*/3);
#else
#pragma unroll
    for (int i = 0; i < 4; ++i) {
      const int id  = t + i * 256;                    // 1024 8-byte chunks
      const int row = id >> 3;
      const int seg = id & 7;
      const unsigned long long vv = *(const unsigned long long*)
          (Aws + ((size_t)(b0 * HH + h) * SS + (s0 + row)) * HDIM + hk + seg * 4);
      *(unsigned long long*)&lsA[row * P_ASTR + seg * 4] = vv;
    }
#endif
#pragma unroll
    for (int i = 0; i < 4; ++i) {
      const int id   = t + i * 256;
      const int krow = id >> 5;
      const int ncg  = id & 31;
      const float4 f = *(const float4*)(W + (size_t)(kk + krow) * DD + nb + ncg * 4);
      const unsigned p01 = f2bf_pk(f.x, f.y);
      const unsigned p23 = f2bf_pk(f.z, f.w);
      lsB[(ncg * 4 + 0) * P_BSTR + krow] = (unsigned short)p01;
      lsB[(ncg * 4 + 1) * P_BSTR + krow] = (unsigned short)(p01 >> 16);
      lsB[(ncg * 4 + 2) * P_BSTR + krow] = (unsigned short)p23;
      lsB[(ncg * 4 + 3) * P_BSTR + krow] = (unsigned short)(p23 >> 16);
    }
#if USE_TDM
    if (wave == 0) __builtin_amdgcn_s_wait_tensorcnt(0);
#endif
    __syncthreads();

    v16bf af[2], bfr[4];
#pragma unroll
    for (int fm = 0; fm < 2; ++fm)
      af[fm] = load_fragA(&lsA[(wm * 32 + fm * 16 + r) * P_ASTR], half);
#pragma unroll
    for (int fn = 0; fn < 4; ++fn)
      bfr[fn] = load_fragB(&lsB[(wn * 64 + fn * 16 + r) * P_BSTR], half);
#pragma unroll
    for (int fm = 0; fm < 2; ++fm)
#pragma unroll
      for (int fn = 0; fn < 4; ++fn)
        acc[fm][fn] = wmma_bf16(af[fm], bfr[fn], acc[fm][fn]);
  }

#pragma unroll
  for (int fm = 0; fm < 2; ++fm) {
#pragma unroll
    for (int fn = 0; fn < 4; ++fn) {
      const int col = nb + wn * 64 + fn * 16 + r;
      const float bc = bias[col];
#pragma unroll
      for (int j = 0; j < 8; ++j) {
        const int row = mb + wm * 32 + fm * 16 + half * 8 + j;
        out[(size_t)row * DD + col] = acc[fm][fn][j] + bc;
      }
    }
  }
}

// ---------------------------------------------------------------------------
// Flash attention: grid (S/64, B*H); 128 threads (4 waves x 16 q-rows).
// Scale 1/sqrt(HD) folded into Q. Causal mask applied analytically.
// ---------------------------------------------------------------------------
#define F_KSTR 72   // padded row stride (144B, 16B aligned)
#define F_PSTR 72

__global__ void __launch_bounds__(128) flash_attn_kernel(
    const unsigned short* __restrict__ Q, const unsigned short* __restrict__ K,
    const unsigned short* __restrict__ V, unsigned short* __restrict__ O)
{
  __shared__ unsigned short lsK[64 * F_KSTR];      // [key][dim]
  __shared__ unsigned short lsV[64 * F_KSTR];      // [dim][key] (transposed)
  __shared__ unsigned short lsP[4 * 16 * F_PSTR];  // per-wave P scratch

  const int t    = threadIdx.x;
  const int qb   = blockIdx.x;       // q-block of 64 rows
  const int bh   = blockIdx.y;       // fused (b,h)
  const int wave = t >> 5, lane = t & 31;
  const int half = lane >> 4, r = lane & 15;

  const size_t headBase = (size_t)bh * SS * HDIM;

  // Persistent Q fragments (16x64 per wave -> two 16x32 A-frags)
  const int qrow = qb * 64 + wave * 16 + r;
  const unsigned short* qp = Q + headBase + (size_t)qrow * HDIM;
  const v16bf aq0 = load_fragA(qp, half);
  const v16bf aq1 = load_fragA(qp + 32, half);

  v8f oacc[4] = {};
  float mrow[8], lrow[8];
#pragma unroll
  for (int j = 0; j < 8; ++j) { mrow[j] = -3.0e38f; lrow[j] = 0.f; }

  unsigned short* myP = &lsP[wave * 16 * F_PSTR];

  for (int jb = 0; jb <= qb; ++jb) {
    const int kb = jb * 64;
    __syncthreads();
#if USE_TDM
    // K tile (64x64 bf16, natural layout) via TDM; hw pads rows to stride 72.
    if (wave == 0)
      tdm_load_2d_bf16(lsK, K + headBase + (size_t)kb * HDIM,
                       /*tile_w=*/64, /*tile_h=*/64, /*row_stride=*/HDIM,
                       /*pad_interval(32dw)=*/4, /*pad_amount(4dw)=*/3);
    // V tile staged transposed by all threads (TDM cannot transpose).
#pragma unroll
    for (int i = 0; i < 8; ++i) {
      const int id  = t + i * 128;
      const int key = id >> 4;
      const int seg = id & 15;
      const unsigned long long vv = *(const unsigned long long*)
          (V + headBase + (size_t)(kb + key) * HDIM + seg * 4);
      const int d = seg * 4;
      lsV[(d + 0) * F_KSTR + key] = (unsigned short)(vv);
      lsV[(d + 1) * F_KSTR + key] = (unsigned short)(vv >> 16);
      lsV[(d + 2) * F_KSTR + key] = (unsigned short)(vv >> 32);
      lsV[(d + 3) * F_KSTR + key] = (unsigned short)(vv >> 48);
    }
    if (wave == 0) __builtin_amdgcn_s_wait_tensorcnt(0);
#else
#pragma unroll
    for (int i = 0; i < 8; ++i) {
      const int id  = t + i * 128;
      const int key = id >> 4;
      const int seg = id & 15;
      const size_t goff = headBase + (size_t)(kb + key) * HDIM + seg * 4;
      const unsigned long long kv = *(const unsigned long long*)(K + goff);
      *(unsigned long long*)&lsK[key * F_KSTR + seg * 4] = kv;
      const unsigned long long vv = *(const unsigned long long*)(V + goff);
      const int d = seg * 4;
      lsV[(d + 0) * F_KSTR + key] = (unsigned short)(vv);
      lsV[(d + 1) * F_KSTR + key] = (unsigned short)(vv >> 16);
      lsV[(d + 2) * F_KSTR + key] = (unsigned short)(vv >> 32);
      lsV[(d + 3) * F_KSTR + key] = (unsigned short)(vv >> 48);
    }
#endif
    __syncthreads();

    if (jb < qb) {  // pull next K/V tile toward L2 (global_prefetch_b8)
      const size_t noff = headBase + (size_t)(kb + 64 + (t & 63)) * HDIM;
      __builtin_prefetch(K + noff, 0, 1);
      __builtin_prefetch(V + noff, 0, 1);
    }

    // scores S = Q @ K^T : 16x64 per wave (4 n-frags x 2 k-steps)
    v8f sc[4];
#pragma unroll
    for (int fn = 0; fn < 4; ++fn) {
      const unsigned short* kp = &lsK[(fn * 16 + r) * F_KSTR];
      const v16bf b0 = load_fragB(kp, half);
      const v16bf b1 = load_fragB(kp + 32, half);
      v8f z = {};
      z = wmma_bf16(aq0, b0, z);
      z = wmma_bf16(aq1, b1, z);
      sc[fn] = z;
    }

    if (jb == qb) {  // causal mask on diagonal block only
#pragma unroll
      for (int fn = 0; fn < 4; ++fn) {
        const int kidx = kb + fn * 16 + r;
#pragma unroll
        for (int j = 0; j < 8; ++j) {
          const int qidx = qb * 64 + wave * 16 + half * 8 + j;
          if (kidx > qidx) sc[fn][j] = -3.0e38f;
        }
      }
    }

    // Online softmax. C-layout: row = half*8+j (16 lanes/row -> xor 8..1)
#pragma unroll
    for (int j = 0; j < 8; ++j) {
      float mj = fmaxf(fmaxf(sc[0][j], sc[1][j]), fmaxf(sc[2][j], sc[3][j]));
#pragma unroll
      for (int off = 8; off >= 1; off >>= 1) mj = fmaxf(mj, __shfl_xor(mj, off, 32));
      const float mnew = fmaxf(mrow[j], mj);
      const float corr = __expf(mrow[j] - mnew);
      float rs = 0.f;
#pragma unroll
      for (int fn = 0; fn < 4; ++fn) {
        const float p = __expf(sc[fn][j] - mnew);
        sc[fn][j] = p;
        rs += p;
      }
#pragma unroll
      for (int off = 8; off >= 1; off >>= 1) rs += __shfl_xor(rs, off, 32);
      lrow[j] = lrow[j] * corr + rs;
      mrow[j] = mnew;
#pragma unroll
      for (int fd = 0; fd < 4; ++fd) oacc[fd][j] *= corr;
    }

    // P: C-layout -> LDS -> A-layout (wave-local; hw waits via DScnt)
#pragma unroll
    for (int fn = 0; fn < 4; ++fn)
#pragma unroll
      for (int j = 0; j < 8; ++j)
        myP[(half * 8 + j) * F_PSTR + fn * 16 + r] = f2bf(sc[fn][j]);

    const unsigned short* pp = &myP[r * F_PSTR];
    const v16bf ap0 = load_fragA(pp, half);
    const v16bf ap1 = load_fragA(pp + 32, half);

    // O += P @ V : k-dim = keys; lsV is [dim][key] so columns are contiguous
#pragma unroll
    for (int fd = 0; fd < 4; ++fd) {
      const unsigned short* vp = &lsV[(fd * 16 + r) * F_KSTR];
      const v16bf b0 = load_fragB(vp, half);
      const v16bf b1 = load_fragB(vp + 32, half);
      oacc[fd] = wmma_bf16(ap0, b0, oacc[fd]);
      oacc[fd] = wmma_bf16(ap1, b1, oacc[fd]);
    }
  }

  // Finalize: O /= l, write bf16 head-major
#pragma unroll
  for (int j = 0; j < 8; ++j) {
    const float inv = 1.0f / lrow[j];
    const int row = qb * 64 + wave * 16 + half * 8 + j;
    unsigned short* op = O + headBase + (size_t)row * HDIM;
#pragma unroll
    for (int fd = 0; fd < 4; ++fd)
      op[fd * 16 + r] = f2bf(oacc[fd][j] * inv);
  }
}

// ---------------------------------------------------------------------------
extern "C" void kernel_launch(void* const* d_in, const int* in_sizes, int n_in,
                              void* d_out, int out_size, void* d_ws, size_t ws_size,
                              hipStream_t stream) {
  (void)in_sizes; (void)n_in; (void)out_size; (void)ws_size;
  const float* q  = (const float*)d_in[0];
  const float* k  = (const float*)d_in[1];
  const float* v  = (const float*)d_in[2];
  // d_in[3] is the causal mask; applied analytically in flash_attn_kernel.
  const float* Wq = (const float*)d_in[4];
  const float* bq = (const float*)d_in[5];
  const float* Wk = (const float*)d_in[6];
  const float* bk = (const float*)d_in[7];
  const float* Wv = (const float*)d_in[8];
  const float* bv = (const float*)d_in[9];
  const float* Wo = (const float*)d_in[10];
  const float* bo = (const float*)d_in[11];
  float* out = (float*)d_out;

  const size_t headElems = (size_t)BB * HH * SS * HDIM;  // 8.39M bf16 each
  unsigned short* Qw = (unsigned short*)d_ws;
  unsigned short* Kw = Qw + headElems;
  unsigned short* Vw = Kw + headElems;
  unsigned short* Ow = Vw + headElems;

  const dim3 gProj((BB * SS) / P_TM, DD / P_TN);  // (64, 8)
  qkv_proj_kernel<<<gProj, 256, 0, stream>>>(q, Wq, bq, Qw, 0.125f);  // 1/sqrt(64)
  qkv_proj_kernel<<<gProj, 256, 0, stream>>>(k, Wk, bk, Kw, 1.0f);
  qkv_proj_kernel<<<gProj, 256, 0, stream>>>(v, Wv, bv, Vw, 1.0f);

  flash_attn_kernel<<<dim3(SS / 64, BB * HH), 128, 0, stream>>>(Qw, Kw, Vw, Ow);

  out_proj_kernel<<<gProj, 256, 0, stream>>>(Ow, Wo, bo, out);
}
